// GenerateNodes_54494545051886
// MI455X (gfx1250) — matrix operand order repaired and checked by reading, not compiled, not run on hardware
//
#include <hip/hip_runtime.h>

// GEMM view of the reference:
//   out[(n*256+t)*192 + col] = sum_k mix[n,t,k] * W[(j*3+c)*16576 + k] + b[j*3+c]
//   col = c*64 + j,  mix = concat( x transposed to (n,t,c_in*64+v), seeds (n,t,192) )
//
// bf16 WMMA (v_wmma_f32_16x16x32_bf16), fp32 accumulate.
// Roofline: 26 GFLOP vs 287 MB => memory-bound only on the bf16 matrix path;
// target ~12 us @ 23.3 TB/s. Double-buffered LDS A+B slabs, one barrier/step;
// all global loads are scalar-base + invariant 32-bit voffset (SADDR form),
// K-loop split x-region / boundary / seeds-region so the hot loop is branch-free.

typedef __attribute__((ext_vector_type(16))) __bf16 v16bf;
typedef __attribute__((ext_vector_type(2)))  __bf16 v2bf;
typedef __attribute__((ext_vector_type(8)))  float  v8f;

#define KX      16384          // x portion of K
#define KS      192            // seeds portion of K
#define IN_DIM  16576
#define LDST    40             // LDS row stride in halves (80 B: 16B-aligned, bank-padded)

union Frag16 { v16bf v; uint4 q[2]; };

static __device__ __forceinline__ unsigned pk2(float lo, float hi) {
  v2bf v; v[0] = (__bf16)lo; v[1] = (__bf16)hi;   // -> v_cvt_pk_bf16_f32
  return __builtin_bit_cast(unsigned, v);
}
static __device__ __forceinline__ uint4 pk8(const float4 a, const float4 b) {
  uint4 q;
  q.x = pk2(a.x, a.y); q.y = pk2(a.z, a.w);
  q.z = pk2(b.x, b.y); q.w = pk2(b.z, b.w);
  return q;
}

__global__ __launch_bounds__(256)
void GenerateNodes_gemm_bf16wmma(const float* __restrict__ x,
                                 const float* __restrict__ seeds,
                                 const float* __restrict__ W,
                                 const float* __restrict__ b,
                                 float* __restrict__ out) {
  __shared__ __bf16 ldsA[2][64 * LDST];   // A slab: rows = t, 32 k, bf16 (2x5120 B)
  __shared__ __bf16 ldsB[2][64 * LDST];   // B slab: rows = block-local col

  const int tid  = threadIdx.x;
  const int lane = tid & 31;
  const int wave = tid >> 5;              // 0..7
  const int bm   = blockIdx.x / 3;        // 0..63  -> 64-row M tile
  const int bn   = blockIdx.x % 3;        // 0..2   -> 64-col N chunk
  const int n    = bm >> 2;
  const int t0   = (bm & 3) * 64;

  const int nt_l = wave >> 1;             // 0..3  N-tile within chunk
  const int mh   = wave & 1;              // 0..1  32-row M half
  const int lcol = nt_l * 16 + (lane & 15);     // block-local col 0..63
  const int col  = bn * 64 + lcol;              // global col 0..191
  const int j    = col & 63;
  const int cidx = col >> 6;
  const int g    = lane >> 4;             // lane-half selects K sub-range

  const float bias = b[j * 3 + cidx];

  // cooperative loader: thread -> (row lr 0..63, k-offset lk in {0,8,16,24})
  const int lr = tid >> 2;
  const int lk = (tid & 3) * 8;

  // scalar-uniform bases; per-thread parts are loop-invariant 32-bit offsets
  const float* __restrict__ xu = x + (size_t)n * (256u * 256u * 64u) + (size_t)t0 * 64;
  const float* __restrict__ su = seeds + ((size_t)(n * 256 + t0)) * KS;
  const int offAx = lr * 64 + lk;                          // x:     t-stride 64
  const int offAs = lr * KS + lk;                          // seeds: t-stride 192
  const int wcol  = bn * 64 + lr;
  const int offW  = ((wcol & 63) * 3 + (wcol >> 6)) * IN_DIM + lk;   // < 2^22

  // LDS store/load invariant offsets
  const int stoff = lr * LDST + lk;
  const int boff  = lcol * LDST + g * 8;
  const int r0    = mh * 32 + (lane & 15);
  const int aoff0 = r0 * LDST + g * 8;
  const int aoff1 = (r0 + 16) * LDST + g * 8;

  v8f acc[2] = {v8f{}, v8f{}};
  float4 a0, a1, b0, b1;

  // ---- one pipelined step: compute slab in `buf`, stage prefetched regs to buf^1
  auto compute_stage = [&](const int buf) {
    const __bf16* __restrict__ la = ldsA[buf];
    const __bf16* __restrict__ lb = ldsB[buf];
    Frag16 bfr, af0, af1;
    // B (32x16 bf16): VGPR0-3 = K g*8..g*8+7, VGPR4-7 = +16 (same as A)
    bfr.q[0] = *(const uint4*)(lb + boff);
    bfr.q[1] = *(const uint4*)(lb + boff + 16);
    // A (16x32 bf16): VGPR0-3 = K g*8..g*8+7, VGPR4-7 = +16
    af0.q[0] = *(const uint4*)(la + aoff0);
    af0.q[1] = *(const uint4*)(la + aoff0 + 16);
    af1.q[0] = *(const uint4*)(la + aoff1);
    af1.q[1] = *(const uint4*)(la + aoff1 + 16);
    __builtin_amdgcn_sched_group_barrier(0x100, 6, 0);   // group: 6 DS reads
    acc[0] = __builtin_amdgcn_wmma_f32_16x16x32_bf16(
        false, af0.v, false, bfr.v, (short)0, acc[0], false, false);
    acc[1] = __builtin_amdgcn_wmma_f32_16x16x32_bf16(
        false, af1.v, false, bfr.v, (short)0, acc[1], false, false);
    __builtin_amdgcn_sched_group_barrier(0x008, 2, 0);   // then: 2 WMMAs
    // convert + store prefetched slab (loadcnt waits land after the WMMAs)
    *(uint4*)(&ldsA[buf ^ 1][stoff]) = pk8(a0, a1);
    *(uint4*)(&ldsB[buf ^ 1][stoff]) = pk8(b0, b1);
    __syncthreads();
  };

  auto pf_x = [&](const int kn) {        // prefetch x-region slab (c_in = kn>>6)
    const float* sa = xu + ((kn >> 6) << 14) + (kn & 32) + offAx;
    a0 = *(const float4*)(sa); a1 = *(const float4*)(sa + 4);
    const float* sw = W + kn + offW;
    b0 = *(const float4*)(sw); b1 = *(const float4*)(sw + 4);
  };
  auto pf_s = [&](const int kn) {        // prefetch seeds-region slab
    const float* sa = su + (kn - KX) + offAs;
    a0 = *(const float4*)(sa); a1 = *(const float4*)(sa + 4);
    const float* sw = W + kn + offW;
    b0 = *(const float4*)(sw); b1 = *(const float4*)(sw + 4);
  };

  // ---- prologue: stage slab 0 into buffer 0 ----
  pf_x(0);
  *(uint4*)(&ldsA[0][stoff]) = pk8(a0, a1);
  *(uint4*)(&ldsB[0][stoff]) = pk8(b0, b1);
  __syncthreads();

  // ---- x phase: s = 0..509, unrolled x2 so buffer parity is compile-time ----
  for (int s = 0; s < 510; s += 2) {
    pf_x((s + 1) * 32); compute_stage(0);
    pf_x((s + 2) * 32); compute_stage(1);
  }
  // ---- boundary: s=510 (x), s=511 (x compute, seeds prefetch) ----
  pf_x(511 * 32); compute_stage(0);
  pf_s(KX);       compute_stage(1);      // stages seeds slab 512 -> buf 0
  // ---- seeds phase: s = 512..517 (last prefetch clamped -> dead store) ----
  pf_s(513 * 32); compute_stage(0);
  pf_s(514 * 32); compute_stage(1);
  pf_s(515 * 32); compute_stage(0);
  pf_s(516 * 32); compute_stage(1);
  pf_s(517 * 32); compute_stage(0);
  pf_s(517 * 32); compute_stage(1);

  // ---- epilogue: C layout VGPR v -> row v + 8*g, col = lane&15 ----
  const size_t obase = (size_t)n * 256 * 192;
#pragma unroll
  for (int ms = 0; ms < 2; ++ms) {
#pragma unroll
    for (int vg = 0; vg < 8; ++vg) {
      const int t = t0 + mh * 32 + ms * 16 + vg + 8 * g;
      out[obase + (size_t)t * 192 + col] = acc[ms][vg] + bias;
    }
  }
}

extern "C" void kernel_launch(void* const* d_in, const int* in_sizes, int n_in,
                              void* d_out, int out_size, void* d_ws, size_t ws_size,
                              hipStream_t stream) {
  const float* x     = (const float*)d_in[0];  // (16,256,256,64)
  const float* seeds = (const float*)d_in[1];  // (16,256,3,64)
  const float* W     = (const float*)d_in[2];  // (64,3,16576)
  const float* b     = (const float*)d_in[3];  // (64,3)
  float* out         = (float*)d_out;          // (16,256,3,64)
  (void)in_sizes; (void)n_in; (void)out_size; (void)d_ws; (void)ws_size;

  dim3 grid(192);   // 64 M-tiles * 3 N-chunks
  dim3 block(256);  // 8 wave32
  GenerateNodes_gemm_bf16wmma<<<grid, block, 0, stream>>>(x, seeds, W, b, out);
}
